// Paper2506_15841v2_MemorySystem_8555574854154
// MI455X (gfx1250) — matrix operand order repaired and checked by reading, not compiled, not run on hardware
//
#include <hip/hip_runtime.h>
#include <hip/hip_bf16.h>

#if __has_include(<hip/amd_detail/amd_gfx1250_TDM.h>)
#define TDM_6ARG 1
#endif

// ---------------------------------------------------------------------------
// Types
// ---------------------------------------------------------------------------
typedef __attribute__((ext_vector_type(16))) __bf16        v16bf;
typedef __attribute__((ext_vector_type(8)))  float         v8f;
typedef __attribute__((ext_vector_type(4)))  unsigned int  u32x4;
typedef __attribute__((ext_vector_type(8)))  int           i32x8;
typedef __attribute__((ext_vector_type(4)))  int           i32x4;

union Frag { v16bf v; u32x4 u[2]; };

// GEMM tiling: WG = 256 threads = 8 wave32. WG tile = 32 (M) x 128 (N);
// wave w owns N strip w*16 and BOTH 16-row M subtiles (2 accumulators),
// so B fragments are reused across 2 C tiles. K staged in BK=128 chunks by
// the Tensor Data Mover with ping-pong LDS double buffering.
#define BK 128
#define OFF_AH 0u               // A_hi: 32*128*2 = 8192
#define OFF_AL 8192u            // A_lo
#define OFF_BH 16384u           // B_hi: 128*128*2 = 32768
#define OFF_BL 49152u           // B_lo
#define STAGE_BYTES 81920u      // one K-stage
#define SMEM_BYTES  163840u     // double buffered

// ---------------------------------------------------------------------------
// Tensor Data Mover: 2D tile (bf16) global -> LDS.  D# per CDNA5 ISA §8.3/8.4:
//   group0: [1:0]=count=1 | lds_addr | global_addr[56:0] | type=2
//   group1: data_size=1 (2B), tensor_dim0/1, tile_dim0/1, tensor_dim0_stride
// ---------------------------------------------------------------------------
__device__ __forceinline__ void tdm_load_bf16_2d(unsigned ldsOff, const void* gp,
                                                 unsigned tileD0, unsigned tileD1,
                                                 unsigned tenD0, unsigned tenD1,
                                                 unsigned strideElems)
{
  unsigned long long ga = (unsigned long long)gp;
  u32x4 g0 = { 1u,                                   // count=1 (user descriptor)
               ldsOff,                               // lds_addr (bytes)
               (unsigned)(ga & 0xFFFFFFFFull),       // global_addr[31:0]
               (unsigned)((ga >> 32) & 0x1FFFFFFull) | 0x80000000u }; // [56:32] | type=2
  i32x8 g1 = { (int)0x10000u,                                        // data_size=1 (2B)
               (int)((tenD0 & 0xFFFFu) << 16),                       // tensor_dim0[15:0]
               (int)((tenD0 >> 16) | ((tenD1 & 0xFFFFu) << 16)),     // d0 hi | d1 lo
               (int)((tenD1 >> 16) | (tileD0 << 16)),                // d1 hi | tile_dim0
               (int)(tileD1 & 0xFFFFu),                              // tile_dim1 (tile_dim2=0)
               (int)strideElems,                                     // tensor_dim0_stride lo
               0, 0 };
  i32x4 gz = { 0, 0, 0, 0 };
#ifdef TDM_6ARG
  i32x8 gz8 = { 0, 0, 0, 0, 0, 0, 0, 0 };
  __builtin_amdgcn_tensor_load_to_lds(g0, g1, gz, gz, gz8, 0);
#else
  __builtin_amdgcn_tensor_load_to_lds(g0, g1, gz, gz, 0);
#endif
}

// ---------------------------------------------------------------------------
// LDS fragment loads (ISA §7.12.2 16-bit layouts), each chunk is a ds_load_b128
// A 16x32: lane half g=l>>4, m=l&15: K = {g*8..g*8+7, 16+g*8..16+g*8+7}
// B 32x16: n=l&15, g=l>>4: K = {g*8..g*8+7, 16+g*8..16+g*8+7} of column n
// ---------------------------------------------------------------------------
__device__ __forceinline__ v16bf frag_a(const __bf16* As, int mrow, int kk, int grp) {
  Frag f;
  const __bf16* p = As + mrow * BK + kk * 32 + grp * 8;
  f.u[0] = *(const u32x4*)p;
  f.u[1] = *(const u32x4*)(p + 16);
  return f.v;
}
__device__ __forceinline__ v16bf frag_b(const __bf16* Bs, int nrow, int kk, int khalf) {
  Frag f;
  const __bf16* p = Bs + nrow * BK + kk * 32 + khalf / 2;
  f.u[0] = *(const u32x4*)p;
  f.u[1] = *(const u32x4*)(p + 16);
  return f.v;
}

#define WMMA_BF16(A, B, C) \
  __builtin_amdgcn_wmma_f32_16x16x32_bf16(false, (A), false, (B), (short)0, (C), false, false)

// ---------------------------------------------------------------------------
// C = A * B^T with A,B given as bf16 (hi,lo) splits of fp32 operands.
// 3-pass compensation: Ah*Bh + Ah*Bl + Al*Bh.
// MODE 0: write fp32 * scale.   MODE 1: write (+bias) re-split into bf16 hi/lo.
// ---------------------------------------------------------------------------
template <int MODE>
__global__ __launch_bounds__(256)
void gemm_bf16x3_kernel(const __bf16* __restrict__ Ah, const __bf16* __restrict__ Al,
                        const __bf16* __restrict__ Bh, const __bf16* __restrict__ Bl,
                        float* __restrict__ Cf, __bf16* __restrict__ Ch,
                        __bf16* __restrict__ Cl, const float* __restrict__ bias,
                        int M, int N, int K, float scale)
{
  extern __shared__ char smem[];

  const int tid   = threadIdx.x;
  const int wave  = tid >> 5;
  const int lane  = tid & 31;
  const int grp   = lane >> 4;
  const int m     = lane & 15;
  const int khalf = grp * 16;
  const int mBase = blockIdx.y * 32;
  const int nBlk  = blockIdx.x * 128;
  const int nRow  = wave * 16 + m;      // this lane's B row within the 128-row tile

  v8f c0 = { 0.f, 0.f, 0.f, 0.f, 0.f, 0.f, 0.f, 0.f };
  v8f c1 = { 0.f, 0.f, 0.f, 0.f, 0.f, 0.f, 0.f, 0.f };

  const int nst = K / BK;

  if (wave == 0) {                       // prologue: stage 0 into buffer 0
    tdm_load_bf16_2d(OFF_AH, Ah + (size_t)mBase * K, BK, 32,  (unsigned)K, (unsigned)M, (unsigned)K);
    tdm_load_bf16_2d(OFF_AL, Al + (size_t)mBase * K, BK, 32,  (unsigned)K, (unsigned)M, (unsigned)K);
    tdm_load_bf16_2d(OFF_BH, Bh + (size_t)nBlk  * K, BK, 128, (unsigned)K, (unsigned)N, (unsigned)K);
    tdm_load_bf16_2d(OFF_BL, Bl + (size_t)nBlk  * K, BK, 128, (unsigned)K, (unsigned)N, (unsigned)K);
  }

  for (int s = 0; s < nst; ++s) {
    const unsigned buf = (unsigned)(s & 1) * STAGE_BYTES;
    if (wave == 0) {
      if (s + 1 < nst) {
        // issue next stage into the other buffer, then wait only for stage s
        // (TDM completes in-order per wave: 8 outstanding -> wait <=4)
        const int kb = (s + 1) * BK;
        const unsigned nb = (unsigned)((s + 1) & 1) * STAGE_BYTES;
        tdm_load_bf16_2d(nb + OFF_AH, Ah + (size_t)mBase * K + kb, BK, 32,  (unsigned)K, (unsigned)M, (unsigned)K);
        tdm_load_bf16_2d(nb + OFF_AL, Al + (size_t)mBase * K + kb, BK, 32,  (unsigned)K, (unsigned)M, (unsigned)K);
        tdm_load_bf16_2d(nb + OFF_BH, Bh + (size_t)nBlk  * K + kb, BK, 128, (unsigned)K, (unsigned)N, (unsigned)K);
        tdm_load_bf16_2d(nb + OFF_BL, Bl + (size_t)nBlk  * K + kb, BK, 128, (unsigned)K, (unsigned)N, (unsigned)K);
        __builtin_amdgcn_s_wait_tensorcnt(4);
      } else {
        __builtin_amdgcn_s_wait_tensorcnt(0);
      }
    }
    __syncthreads();                     // publish stage s LDS tiles to all waves

    const __bf16* As_h = (const __bf16*)(smem + buf + OFF_AH);
    const __bf16* As_l = (const __bf16*)(smem + buf + OFF_AL);
    const __bf16* Bs_h = (const __bf16*)(smem + buf + OFF_BH);
    const __bf16* Bs_l = (const __bf16*)(smem + buf + OFF_BL);

#pragma unroll
    for (int kk = 0; kk < BK / 32; ++kk) {
      v16bf bh  = frag_b(Bs_h, nRow, kk, khalf);
      v16bf bl  = frag_b(Bs_l, nRow, kk, khalf);
      v16bf ah0 = frag_a(As_h, m,      kk, grp);
      v16bf al0 = frag_a(As_l, m,      kk, grp);
      v16bf ah1 = frag_a(As_h, 16 + m, kk, grp);
      v16bf al1 = frag_a(As_l, 16 + m, kk, grp);
      c0 = WMMA_BF16(ah0, bh, c0);
      c0 = WMMA_BF16(ah0, bl, c0);
      c0 = WMMA_BF16(al0, bh, c0);
      c1 = WMMA_BF16(ah1, bh, c1);
      c1 = WMMA_BF16(ah1, bl, c1);
      c1 = WMMA_BF16(al1, bh, c1);
    }
    __syncthreads();                     // reads done before buffer reuse
  }

  // C layout: VGPR r, lanes0-15 -> M=r, lanes16-31 -> M=r+8; N = lane&15
  const int col    = nBlk + wave * 16 + (lane & 15);
  const int rowOff = (lane >> 4) * 8;
  const float bv   = (MODE == 1) ? bias[col] : 0.f;
#pragma unroll
  for (int r = 0; r < 8; ++r) {
    const int row0 = mBase + rowOff + r;
    const int row1 = row0 + 16;
    float v0 = c0[r] * scale;
    float v1 = c1[r] * scale;
    if (MODE == 0) {
      Cf[(size_t)row0 * N + col] = v0;
      Cf[(size_t)row1 * N + col] = v1;
    } else {
      v0 += bv;
      v1 += bv;
      __bf16 h0 = (__bf16)v0;
      __bf16 h1 = (__bf16)v1;
      Ch[(size_t)row0 * N + col] = h0;
      Cl[(size_t)row0 * N + col] = (__bf16)(v0 - (float)h0);
      Ch[(size_t)row1 * N + col] = h1;
      Cl[(size_t)row1 * N + col] = (__bf16)(v1 - (float)h1);
    }
  }
}

// ---------------------------------------------------------------------------
// fp32 -> bf16 hi + bf16 residual
// ---------------------------------------------------------------------------
__global__ void split_kernel(const float* __restrict__ x, __bf16* __restrict__ hi,
                             __bf16* __restrict__ lo, int n)
{
  int i = blockIdx.x * blockDim.x + threadIdx.x;
  if (i < n) {
    float v = x[i];
    __bf16 h = (__bf16)v;
    hi[i] = h;
    lo[i] = (__bf16)(v - (float)h);
  }
}

// ---------------------------------------------------------------------------
// Per-row: softmax stats + top-k (scores monotone under softmax).
// Row (N fp32) lives in dynamic LDS; iterative argmax with JAX tie-breaking
// (ties -> lowest index).
// ---------------------------------------------------------------------------
__global__ __launch_bounds__(256)
void topk_kernel(const float* __restrict__ S, int rowBase, int N, int k,
                 float* __restrict__ outVals, int* __restrict__ outIdx)
{
  extern __shared__ char smem[];
  float* row = (float*)smem;
  __shared__ float redV[256];
  __shared__ int   redI[256];
  __shared__ float sMax, sSum;

  const int tid = threadIdx.x;
  const float* src = S + (size_t)blockIdx.x * N;
  for (int i = tid; i < N; i += 256) row[i] = src[i];
  __syncthreads();

  float mx = -__builtin_inff();
  for (int i = tid; i < N; i += 256) mx = fmaxf(mx, row[i]);
  redV[tid] = mx; __syncthreads();
  for (int off = 128; off > 0; off >>= 1) {
    if (tid < off) redV[tid] = fmaxf(redV[tid], redV[tid + off]);
    __syncthreads();
  }
  if (tid == 0) sMax = redV[0];
  __syncthreads();
  const float rm = sMax;

  float sum = 0.f;
  for (int i = tid; i < N; i += 256) sum += __expf(row[i] - rm);
  redV[tid] = sum; __syncthreads();
  for (int off = 128; off > 0; off >>= 1) {
    if (tid < off) redV[tid] += redV[tid + off];
    __syncthreads();
  }
  if (tid == 0) sSum = redV[0];
  __syncthreads();
  const float rs = sSum;

  const size_t outBase = (size_t)(rowBase + blockIdx.x) * k;
  for (int j = 0; j < k; ++j) {
    float bv = -__builtin_inff(); int bi = 0x7fffffff;
    for (int i = tid; i < N; i += 256) {
      float v = row[i];
      if (v > bv) { bv = v; bi = i; }          // ascending i -> lowest index on tie
    }
    redV[tid] = bv; redI[tid] = bi; __syncthreads();
    for (int off = 128; off > 0; off >>= 1) {
      if (tid < off) {
        float ov = redV[tid + off]; int oi = redI[tid + off];
        if (ov > redV[tid] || (ov == redV[tid] && oi < redI[tid])) {
          redV[tid] = ov; redI[tid] = oi;
        }
      }
      __syncthreads();
    }
    if (tid == 0) {
      int idx = redI[0];
      outVals[outBase + j] = __expf(redV[0] - rm) / rs;
      outIdx[outBase + j]  = idx;
      row[idx] = -__builtin_inff();
    }
    __syncthreads();
  }
}

// ---------------------------------------------------------------------------
// retrieved[slot] = episodes[idx[slot]]  (exact fp32, 128-bit vector copies)
// ---------------------------------------------------------------------------
__global__ void gather_kernel(const float* __restrict__ eps, const int* __restrict__ idx,
                              float* __restrict__ out, int D)
{
  const size_t slot = blockIdx.x;
  const int e = idx[slot];
  const float4* src = (const float4*)(eps + (size_t)e * D);
  float4* dst = (float4*)(out + slot * D);
  for (int i = threadIdx.x; i < (D >> 2); i += blockDim.x) dst[i] = src[i];
}

// ---------------------------------------------------------------------------
extern "C" void kernel_launch(void* const* d_in, const int* in_sizes, int n_in,
                              void* d_out, int out_size, void* d_ws, size_t ws_size,
                              hipStream_t stream)
{
  const float* query    = (const float*)d_in[0];
  const float* episodes = (const float*)d_in[1];
  const float* W        = (const float*)d_in[2];
  const float* bias     = (const float*)d_in[3];

  const int D  = in_sizes[3];                 // 1024
  const int Bq = in_sizes[0] / D;             // 4096
  const int N  = in_sizes[1] / D;             // 32768
  const int k  = out_size / (Bq * (D + 1));   // 64

  float* outRet = (float*)d_out;                       // [Bq, k, D]
  float* outVal = (float*)d_out + (size_t)Bq * k * D;  // [Bq, k]

  char* ws = (char*)d_ws;
  size_t off = 0;
  auto walloc = [&](size_t bytes) -> void* {
    void* p = (void*)(ws + off);
    off += (bytes + 255) & ~(size_t)255;
    return p;
  };
  __bf16* Qh = (__bf16*)walloc((size_t)Bq * D * 2);
  __bf16* Ql = (__bf16*)walloc((size_t)Bq * D * 2);
  __bf16* Wh = (__bf16*)walloc((size_t)D * D * 2);
  __bf16* Wl = (__bf16*)walloc((size_t)D * D * 2);
  __bf16* Ph = (__bf16*)walloc((size_t)Bq * D * 2);
  __bf16* Pl = (__bf16*)walloc((size_t)Bq * D * 2);
  __bf16* Eh = (__bf16*)walloc((size_t)N * D * 2);
  __bf16* El = (__bf16*)walloc((size_t)N * D * 2);
  int*    Ix = (int*)   walloc((size_t)Bq * k * 4);

  // scores staging slice (rows of [slice, N] fp32), sized to remaining ws;
  // must be a multiple of the 32-row M tile
  size_t avail = (ws_size > off) ? (ws_size - off) : 0;
  int slice = (int)(avail / ((size_t)N * 4));
  slice &= ~31;
  if (slice < 32)  slice = 32;
  if (slice > 512) slice = 512;
  float* SC = (float*)walloc((size_t)slice * N * 4);

  (void)hipFuncSetAttribute((const void*)gemm_bf16x3_kernel<0>,
                            hipFuncAttributeMaxDynamicSharedMemorySize, (int)SMEM_BYTES);
  (void)hipFuncSetAttribute((const void*)gemm_bf16x3_kernel<1>,
                            hipFuncAttributeMaxDynamicSharedMemorySize, (int)SMEM_BYTES);
  (void)hipFuncSetAttribute((const void*)topk_kernel,
                            hipFuncAttributeMaxDynamicSharedMemorySize, N * 4);

  // 1) bf16 hi/lo splits of query, W, episodes
  {
    int n = Bq * D;
    split_kernel<<<(n + 255) / 256, 256, 0, stream>>>(query, Qh, Ql, n);
    n = D * D;
    split_kernel<<<(n + 255) / 256, 256, 0, stream>>>(W, Wh, Wl, n);
    n = N * D;
    split_kernel<<<(n + 255) / 256, 256, 0, stream>>>(episodes, Eh, El, n);
  }

  // 2) projection: P = Q @ W^T + b, re-split to bf16 hi/lo
  {
    dim3 g(D / 128, Bq / 32);
    gemm_bf16x3_kernel<1><<<g, 256, SMEM_BYTES, stream>>>(
        Qh, Ql, Wh, Wl, nullptr, Ph, Pl, bias, Bq, D, D, 1.0f);
  }

  // 3) per-slice: scores = (P @ E^T) / sqrt(D), then per-row softmax stats + top-k
  const float invScale = 1.0f / sqrtf((float)D);
  for (int r0 = 0; r0 < Bq; r0 += slice) {
    int rows = (Bq - r0 < slice) ? (Bq - r0) : slice;
    dim3 g(N / 128, rows / 32);
    gemm_bf16x3_kernel<0><<<g, 256, SMEM_BYTES, stream>>>(
        Ph + (size_t)r0 * D, Pl + (size_t)r0 * D, Eh, El,
        SC, nullptr, nullptr, nullptr, rows, N, D, invScale);
    topk_kernel<<<rows, 256, (size_t)N * 4, stream>>>(SC, r0, N, k, outVal, Ix);
  }

  // 4) gather retrieved episodes (exact fp32 rows)
  gather_kernel<<<Bq * k, 64, 0, stream>>>(episodes, Ix, outRet, D);
}
